// GraphKalmanFilter_28913719837354
// MI455X (gfx1250) — compile-verified, matrix operand order checked by reading, not run on hardware
//
#include <hip/hip_runtime.h>
#include <hip/hip_bf16.h>

// ---------------------------------------------------------------------------
// Problem constants (from the reference)
// ---------------------------------------------------------------------------
#define B_   64
#define N_   256
#define S_   9
#define H_   64
#define DEG_ 16
#define O_   81
#define BN_  16384      // B*N
#define E_   262144     // BN*DEG

typedef __attribute__((ext_vector_type(16))) __bf16 v16bf;
typedef __attribute__((ext_vector_type(8)))  float  v8f;

// ---------------------------------------------------------------------------
// WMMA helpers (CDNA5 wave32, V_WMMA_F32_16X16X32_BF16)
//
// A fragment (16x32 bf16): lane = M row (lane&15); per lane 8 VGPR pairs:
//   k = ((v>>2)<<4) + ((lane>>4)<<3) + ((v&3)<<1) + p      (ISA 7.12.2)
//   -> two contiguous 8-float runs per lane: [kh*8, kh*8+8) and +16.
// B fragment assumed layout-symmetric to A with lane = N column (gfx12-style);
// weights are PRE-PACKED into that exact order so the kernel streams 32B/lane.
// C/D: VGPR i holds M = i + 8*(lane>>4), N = lane&15.
// ---------------------------------------------------------------------------
__device__ __forceinline__ v8f wmma_bf16(v16bf a, v16bf b, v8f c) {
  return __builtin_amdgcn_wmma_f32_16x16x32_bf16(false, a, false, b,
                                                 (short)0, c, false, false);
}

__device__ __forceinline__ v16bf load_wfrag(const __bf16* __restrict__ pw,
                                            int kt, int nt, int Ntiles, int lane) {
  return *(const v16bf*)(pw + ((size_t)((kt * Ntiles + nt) * 32 + lane) * 16));
}

// Branch-free A fragment: requires rows 32B-aligned, lda >= Ktiles*32, pads = 0.
__device__ __forceinline__ v16bf load_afrag(const float* __restrict__ Arow,
                                            int kt, int kh) {
  int base0 = kt * 32 + kh * 8;
  const float4 f0 = *(const float4*)(Arow + base0);
  const float4 f1 = *(const float4*)(Arow + base0 + 4);
  const float4 f2 = *(const float4*)(Arow + base0 + 16);
  const float4 f3 = *(const float4*)(Arow + base0 + 20);
  v16bf a;
  a[0]  = (__bf16)f0.x; a[1]  = (__bf16)f0.y; a[2]  = (__bf16)f0.z; a[3]  = (__bf16)f0.w;
  a[4]  = (__bf16)f1.x; a[5]  = (__bf16)f1.y; a[6]  = (__bf16)f1.z; a[7]  = (__bf16)f1.w;
  a[8]  = (__bf16)f2.x; a[9]  = (__bf16)f2.y; a[10] = (__bf16)f2.z; a[11] = (__bf16)f2.w;
  a[12] = (__bf16)f3.x; a[13] = (__bf16)f3.y; a[14] = (__bf16)f3.z; a[15] = (__bf16)f3.w;
  return a;
}

// ---------------------------------------------------------------------------
// Weight packer: W (K,N) f32 row-major -> bf16 fragment order, zero padded.
// ---------------------------------------------------------------------------
__global__ void pack_w(const float* __restrict__ W, __bf16* __restrict__ out,
                       int K, int N, int Ktiles, int Ntiles) {
  int tid = blockIdx.x * 256 + threadIdx.x;
  int total = Ktiles * Ntiles * 32;
  if (tid >= total) return;
  int lane = tid & 31;
  int frag = tid >> 5;
  int kt = frag / Ntiles, nt = frag - kt * Ntiles;
  int n  = nt * 16 + (lane & 15);
  int kh = lane >> 4;
  __bf16* o = out + (size_t)tid * 16;
#pragma unroll
  for (int v = 0; v < 8; ++v) {
#pragma unroll
    for (int p = 0; p < 2; ++p) {
      int k = kt * 32 + ((v >> 2) << 4) + (kh << 3) + ((v & 3) << 1) + p;
      float f = (k < K && n < N) ? W[(size_t)k * N + n] : 0.0f;
      o[v * 2 + p] = (__bf16)f;
    }
  }
}

// ---------------------------------------------------------------------------
// Generic GEMM: C[M,N](ldc) = act( A[M,Kpad](lda) @ W + bias ), M = BN_.
// One wave per 16x16 tile, double-buffered K loop of bf16 WMMAs.
// All tiles are full (A zero-padded); stores are branch-free and zero-fill
// the pad columns n in [N, Ntiles*16)  (ldc must be >= Ntiles*16 when nt>0).
// act: 0 = identity, 1 = leaky_relu(0.01)
// ---------------------------------------------------------------------------
__global__ __launch_bounds__(256) void gemm_wmma(
    const float* __restrict__ A, int lda,
    const __bf16* __restrict__ PW, const float* __restrict__ bias,
    float* __restrict__ C, int ldc,
    int N, int Ktiles, int Ntiles, int act) {
  int gw   = (blockIdx.x * 256 + threadIdx.x) >> 5;
  int lane = threadIdx.x & 31;
  const int Mtiles = BN_ >> 4;
  if (gw >= Mtiles * Ntiles) return;
  int mt = gw / Ntiles, nt = gw - mt * Ntiles;
  int kh = lane >> 4;
  const float* Arow = A + (size_t)(mt * 16 + (lane & 15)) * lda;

  v8f c = {};
  v16bf a = load_afrag(Arow, 0, kh);
  v16bf b = load_wfrag(PW, 0, nt, Ntiles, lane);
  for (int kt = 0; kt < Ktiles; ++kt) {
    v16bf an = a, bn = b;
    if (kt + 1 < Ktiles) {                     // uniform condition: no exec churn
      an = load_afrag(Arow, kt + 1, kh);
      bn = load_wfrag(PW, kt + 1, nt, Ntiles, lane);
      __builtin_prefetch(Arow + (kt + 1) * 32 + 32, 0, 1);
    }
    c = wmma_bf16(a, b, c);
    a = an; b = bn;
  }

  int n = nt * 16 + (lane & 15);
  float bv = bias ? bias[n < N ? n : N - 1] : 0.0f;
  bool on = n < N;
#pragma unroll
  for (int i = 0; i < 8; ++i) {
    int m = mt * 16 + i + 8 * kh;
    float v = c[i] + bv;
    if (act) v = v > 0.0f ? v : 0.01f * v;
    C[(size_t)m * ldc + n] = on ? v : 0.0f;    // zero-fill pad columns
  }
}

// ---------------------------------------------------------------------------
// Node preprocessing: x_tt1 = F@x11, rin = meas - Hm[i].x_tt1 (diag innov),
// yfeat = [dyi_in, Hm[i]] (pad 32), xfeat = [x11-x12, x11-x22] (pad 32).
// ---------------------------------------------------------------------------
__global__ void node_prep(const float* __restrict__ x11, const float* __restrict__ x12,
                          const float* __restrict__ x22, const float* __restrict__ Fm,
                          const float* __restrict__ Hm, const float* __restrict__ meas,
                          const float* __restrict__ dyin,
                          float* __restrict__ xt1, float* __restrict__ rin,
                          float* __restrict__ yfeat, float* __restrict__ xfeat) {
  int n = blockIdx.x * 256 + threadIdx.x;
  if (n >= BN_) return;
  int i = n & (N_ - 1);
  float x[S_], xt[S_];
#pragma unroll
  for (int s = 0; s < S_; ++s) x[s] = x11[n * S_ + s];
#pragma unroll
  for (int s = 0; s < S_; ++s) {
    float a = 0.0f;
#pragma unroll
    for (int t = 0; t < S_; ++t) a += Fm[s * S_ + t] * x[t];
    xt[s] = a;
    xt1[n * S_ + s] = a;
  }
  float yd = 0.0f;
#pragma unroll
  for (int s = 0; s < S_; ++s) yd += Hm[i * S_ + s] * xt[s];
  // rin padded row of 32
  rin[(size_t)n * 32] = meas[n] - yd;
#pragma unroll
  for (int s = 1; s < 32; ++s) rin[(size_t)n * 32 + s] = 0.0f;
  // yfeat padded row of 32
  yfeat[(size_t)n * 32] = dyin[n];
#pragma unroll
  for (int s = 0; s < S_; ++s) yfeat[(size_t)n * 32 + 1 + s] = Hm[i * S_ + s];
#pragma unroll
  for (int s = 10; s < 32; ++s) yfeat[(size_t)n * 32 + s] = 0.0f;
  // xfeat padded row of 32
#pragma unroll
  for (int s = 0; s < S_; ++s) {
    xfeat[(size_t)n * 32 + s]     = x[s] - x12[n * S_ + s];
    xfeat[(size_t)n * 32 + 9 + s] = x[s] - x22[n * S_ + s];
  }
#pragma unroll
  for (int s = 18; s < 32; ++s) xfeat[(size_t)n * 32 + s] = 0.0f;
}

// ---------------------------------------------------------------------------
// GRU gate combine (gates already contain biases): out = (1-z)*tanh(in+r*hn)+z*h
// ---------------------------------------------------------------------------
__global__ void gru_combine(const float* __restrict__ gi, const float* __restrict__ gh,
                            const float* __restrict__ hprev, int hld,
                            float* __restrict__ out, float* __restrict__ out2,
                            int HD, int ldg, int ld2, int off2, int total) {
  int t = blockIdx.x * 256 + threadIdx.x;
  if (t >= total) return;
  int n = t / HD, h = t - n * HD;
  const float* pi = gi + (size_t)n * ldg;
  const float* ph = gh + (size_t)n * ldg;
  float r  = 1.0f / (1.0f + expf(-(pi[h]      + ph[h])));
  float z  = 1.0f / (1.0f + expf(-(pi[HD + h] + ph[HD + h])));
  float nn = tanhf(pi[2 * HD + h] + r * ph[2 * HD + h]);
  float o  = (1.0f - z) * nn + z * hprev[(size_t)n * hld + h];
  out[(size_t)n * HD + h] = o;
  if (out2) out2[(size_t)n * ld2 + off2 + h] = o;
}

// GCN mean-aggregation with uniform degree 17 (16 in-edges + self loop).
__global__ void gcn_agg(const float* __restrict__ xw, const int* __restrict__ src,
                        const float* __restrict__ bg, float* __restrict__ nk) {
  int t = blockIdx.x * 256 + threadIdx.x;
  if (t >= BN_ * 64) return;
  int node = t >> 6, col = t & 63;
  const int* sp = src + node * DEG_;
  float s = xw[(size_t)node * 64 + col];
#pragma unroll
  for (int j = 0; j < DEG_; ++j) s += xw[(size_t)sp[j] * 64 + col];
  nk[(size_t)node * 160 + col] = s * (1.0f / 289.0f) + bg[col];
}

// pred_sigma -> nk[:, 64:145], zero-fill nk[:, 145:160]
__global__ void copy_ps(const float* __restrict__ ps, float* __restrict__ nk) {
  int t = blockIdx.x * 256 + threadIdx.x;
  if (t >= BN_ * 96) return;
  int n = t / 96, j = t - n * 96;
  nk[(size_t)n * 160 + 64 + j] = (j < O_) ? ps[(size_t)n * O_ + j] : 0.0f;
}

// phi = x_tt1 + K @ cross  (K = node_kalman reshaped 9x9, row stride 96)
__global__ void phi_kernel(const float* __restrict__ xt1, const float* __restrict__ nodeK,
                           const float* __restrict__ cross, float* __restrict__ phi) {
  int n = blockIdx.x * 256 + threadIdx.x;
  if (n >= BN_) return;
  float cr[S_];
#pragma unroll
  for (int j = 0; j < S_; ++j) cr[j] = cross[n * S_ + j];
#pragma unroll
  for (int s = 0; s < S_; ++s) {
    float a = xt1[n * S_ + s];
#pragma unroll
    for (int j = 0; j < S_; ++j) a += nodeK[(size_t)n * 96 + s * S_ + j] * cr[j];
    phi[n * S_ + s] = a;
  }
}

// SimpleConv(aggr='mean'): x_tt[t] = mean_j phi[src[16t+j]]
__global__ void xtt_kernel(const float* __restrict__ phi, const int* __restrict__ src,
                           float* __restrict__ o) {
  int t = blockIdx.x * 256 + threadIdx.x;
  if (t >= BN_ * S_) return;
  int node = t / S_, s = t - node * S_;
  const int* sp = src + node * DEG_;
  float acc = 0.0f;
#pragma unroll
  for (int j = 0; j < DEG_; ++j) acc += phi[(size_t)sp[j] * S_ + s];
  o[t] = acc * (1.0f / 16.0f);
}

// ---------------------------------------------------------------------------
// Fused edge MLP (cross_mlp): one wave per node = its 16 edges.
// z(16x18) -> 64 -> 64 -> 64 -> 64 -> 9, times edge_features, column-mean.
// 30 bf16 WMMAs per wave; activations re-staged through a private LDS tile.
// ---------------------------------------------------------------------------
#define LST 72

__device__ __forceinline__ v16bf lds_afrag(float st[16][LST], int lane, int kbase) {
  int row = lane & 15, kh = lane >> 4;
  v16bf a;
#pragma unroll
  for (int v = 0; v < 8; ++v) {
    int k = kbase + ((v >> 2) << 4) + (kh << 3) + ((v & 3) << 1);
    a[2 * v]     = (__bf16)st[row][k];
    a[2 * v + 1] = (__bf16)st[row][k + 1];
  }
  return a;
}

__device__ __forceinline__ void stage_store(float st[16][LST], int lane,
                                            const v8f* acc, const float* __restrict__ bias) {
  int n0 = lane & 15, kh = lane >> 4;
#pragma unroll
  for (int nt = 0; nt < 4; ++nt) {
    int n = nt * 16 + n0;
    float bv = bias[n];
#pragma unroll
    for (int i = 0; i < 8; ++i) {
      float v = acc[nt][i] + bv;
      st[i + 8 * kh][n] = v > 0.0f ? v : 0.01f * v;     // leaky relu
    }
  }
}

__global__ __launch_bounds__(256) void edge_mlp(
    const float* __restrict__ xr, const float* __restrict__ Hm,
    const float* __restrict__ meas, const float* __restrict__ xt1,
    const int* __restrict__ src,
    const __bf16* __restrict__ w1, const float* __restrict__ b1,
    const __bf16* __restrict__ w2, const float* __restrict__ b2,
    const __bf16* __restrict__ w3, const float* __restrict__ b3,
    const __bf16* __restrict__ w4, const float* __restrict__ b4,
    const __bf16* __restrict__ w5, const float* __restrict__ b5,
    float* __restrict__ cross) {
  __shared__ float stage[8][16][LST];
  const int w    = threadIdx.x >> 5;
  const int lane = threadIdx.x & 31;
  const int t    = blockIdx.x * 8 + w;          // node id; grid is exact
  float (*st)[LST] = stage[w];

  // Build z rows + edge_features. edge j of node t: src global id sj,
  // s_l = sj % N, meas[g, s_l] == meas[sj]; ef = meas[sj] - Hm[s_l].x_tt1[t].
  if (lane < 16) {
    int sj = src[t * DEG_ + lane];
    int sl = sj & (N_ - 1);
    float ef = meas[sj];
#pragma unroll
    for (int s = 0; s < S_; ++s) {
      float h = Hm[sl * S_ + s];
      st[lane][9 + s] = h;
      ef -= h * xt1[t * S_ + s];
      st[lane][s] = xr[t * 16 + s];
    }
#pragma unroll
    for (int c = 18; c < 32; ++c) st[lane][c] = 0.0f;
    st[lane][64] = ef;                          // survives re-staging (cols 0..63)
  }
  __syncthreads();

  v8f acc[4];
  const v8f z8 = {};
  // Layer 1: K=18 (1 ktile), N=64
  {
    v16bf a0 = lds_afrag(st, lane, 0);
#pragma unroll
    for (int nt = 0; nt < 4; ++nt)
      acc[nt] = wmma_bf16(a0, load_wfrag(w1, 0, nt, 4, lane), z8);
    __syncthreads();
    stage_store(st, lane, acc, b1);
    __syncthreads();
  }
  // Layers 2..4: K=64 (2 ktiles), N=64
  const __bf16* wmid[3] = {w2, w3, w4};
  const float*  bmid[3] = {b2, b3, b4};
#pragma unroll
  for (int L = 0; L < 3; ++L) {
    v16bf a0 = lds_afrag(st, lane, 0);
    v16bf a1 = lds_afrag(st, lane, 32);
#pragma unroll
    for (int nt = 0; nt < 4; ++nt) {
      v8f c   = wmma_bf16(a0, load_wfrag(wmid[L], 0, nt, 4, lane), z8);
      acc[nt] = wmma_bf16(a1, load_wfrag(wmid[L], 1, nt, 4, lane), c);
    }
    __syncthreads();
    stage_store(st, lane, acc, bmid[L]);
    __syncthreads();
  }
  // Layer 5: K=64, N=9 -> m = (zW+b)*ef, then mean over the 16 edge rows.
  {
    int kh = lane >> 4;
    float efr[8];
#pragma unroll
    for (int i = 0; i < 8; ++i) efr[i] = st[i + 8 * kh][64];
    v16bf a0 = lds_afrag(st, lane, 0);
    v16bf a1 = lds_afrag(st, lane, 32);
    v8f c = wmma_bf16(a0, load_wfrag(w5, 0, 0, 1, lane), z8);
    c     = wmma_bf16(a1, load_wfrag(w5, 1, 0, 1, lane), c);
    int n = lane & 15;
    float bv = (n < S_) ? b5[n] : 0.0f;
    float part = 0.0f;
#pragma unroll
    for (int i = 0; i < 8; ++i) part += (c[i] + bv) * efr[i];
    part += __shfl_xor(part, 16, 32);           // combine row halves
    if (lane < S_) cross[t * S_ + lane] = part * (1.0f / 16.0f);
  }
}

// ---------------------------------------------------------------------------
// Host launcher
// ---------------------------------------------------------------------------
extern "C" void kernel_launch(void* const* d_in, const int* in_sizes, int n_in,
                              void* d_out, int out_size, void* d_ws, size_t ws_size,
                              hipStream_t stream) {
  // ---- inputs (setup_inputs() insertion order, params flattened in order) ----
  const float* Hm    = (const float*)d_in[0];
  const float* Fm    = (const float*)d_in[1];
  const float* dyiW1 = (const float*)d_in[2];  const float* dyiB1 = (const float*)d_in[3];
  const float* dyiW2 = (const float*)d_in[4];  const float* dyiB2 = (const float*)d_in[5];
  const float* rW1   = (const float*)d_in[6];  const float* rB1   = (const float*)d_in[7];
  const float* rW2   = (const float*)d_in[8];  const float* rB2   = (const float*)d_in[9];
  const float* gWih  = (const float*)d_in[10]; const float* gWhh  = (const float*)d_in[11];
  const float* gBih  = (const float*)d_in[12]; const float* gBhh  = (const float*)d_in[13];
  const float* gcnW  = (const float*)d_in[14]; const float* gcnB  = (const float*)d_in[15];
  const float* sigW1 = (const float*)d_in[16]; const float* sigB1 = (const float*)d_in[17];
  const float* sigW2 = (const float*)d_in[18]; const float* sigB2 = (const float*)d_in[19];
  const float* sgWih = (const float*)d_in[20]; const float* sgWhh = (const float*)d_in[21];
  const float* sgBih = (const float*)d_in[22]; const float* sgBhh = (const float*)d_in[23];
  const float* nW1   = (const float*)d_in[24]; const float* nB1   = (const float*)d_in[25];
  const float* nW2   = (const float*)d_in[26]; const float* nB2   = (const float*)d_in[27];
  const float* nW3   = (const float*)d_in[28]; const float* nB3   = (const float*)d_in[29];
  const float* crW   = (const float*)d_in[30]; const float* crB   = (const float*)d_in[31];
  const float* cW1   = (const float*)d_in[32]; const float* cB1   = (const float*)d_in[33];
  const float* cW2   = (const float*)d_in[34]; const float* cB2   = (const float*)d_in[35];
  const float* cW3   = (const float*)d_in[36]; const float* cB3   = (const float*)d_in[37];
  const float* cW4   = (const float*)d_in[38]; const float* cB4   = (const float*)d_in[39];
  const float* cW5   = (const float*)d_in[40]; const float* cB5   = (const float*)d_in[41];
  const float* meas  = (const float*)d_in[42];
  const float* x11   = (const float*)d_in[43];
  const float* x12   = (const float*)d_in[44];
  const float* x22   = (const float*)d_in[45];
  const float* dyin  = (const float*)d_in[46];
  const int*   eidx  = (const int*)d_in[47];   // (2,E); src = first E
  // d_in[48] adj_matrix: dead code in the reference (only feeds unused _y_local)
  const float* hr    = (const float*)d_in[49];
  const float* psig  = (const float*)d_in[50];
  const int*   src   = eidx;

  // ---- outputs: x_tt, x_tt1, edge_index, hidden_r, pred_sigma (flat) ----
  float* out    = (float*)d_out;
  float* o_xtt  = out;
  float* o_xtt1 = out + 147456;
  float* o_eidx = out + 294912;
  float* o_hr   = out + 819200;
  float* o_ps   = out + 1867776;

  // ---- workspace allocator ----
  char* wsb = (char*)d_ws;
  size_t off = 0;
  auto alloc = [&](size_t bytes) -> void* {
    void* p = wsb + off;
    off += bytes;
    off = (off + 255) & ~(size_t)255;
    return p;
  };
  auto packw = [&](const float* W, int K, int N) -> __bf16* {
    int Kt = (K + 31) / 32, Nt = (N + 15) / 16;
    __bf16* p = (__bf16*)alloc((size_t)Kt * Nt * 512 * sizeof(__bf16));
    int total = Kt * Nt * 32;
    pack_w<<<(total + 255) / 256, 256, 0, stream>>>(W, p, K, N, Kt, Nt);
    return p;
  };

  // ---- pack all weights to bf16 fragment order (~0.3 MB, L2-resident) ----
  __bf16* p_dyi1 = packw(dyiW1, 10, 64);
  __bf16* p_dyi2 = packw(dyiW2, 64, 64);
  __bf16* p_r1   = packw(rW1,    1, 64);
  __bf16* p_r2   = packw(rW2,   64, 64);
  __bf16* p_gih  = packw(gWih,  64, 192);
  __bf16* p_ghh  = packw(gWhh,  64, 192);
  __bf16* p_gcn  = packw(gcnW, 128, 64);
  __bf16* p_sig1 = packw(sigW1, 18, 64);
  __bf16* p_sig2 = packw(sigW2, 64, 64);
  __bf16* p_sgih = packw(sgWih, 64, 243);
  __bf16* p_sghh = packw(sgWhh, 81, 243);
  __bf16* p_n1   = packw(nW1,  145, 128);
  __bf16* p_n2   = packw(nW2,  128, 64);
  __bf16* p_n3   = packw(nW3,   64, 81);
  __bf16* p_xr   = packw(crW,   64, 9);
  __bf16* p_c1   = packw(cW1,   18, 64);
  __bf16* p_c2   = packw(cW2,   64, 64);
  __bf16* p_c3   = packw(cW3,   64, 64);
  __bf16* p_c4   = packw(cW4,   64, 64);
  __bf16* p_c5   = packw(cW5,   64, 9);

  // ---- f32 temporaries (row strides padded to full K-tiles, rows 32B aligned)
  float* rin   = (float*)alloc((size_t)BN_ * 32 * 4);
  float* yfeat = (float*)alloc((size_t)BN_ * 32 * 4);
  float* xfeat = (float*)alloc((size_t)BN_ * 32 * 4);
  float* gnn   = (float*)alloc((size_t)BN_ * 128 * 4);   // [dyi | r_out]; reused as fc_node hidden
  float* t64b  = (float*)alloc((size_t)BN_ * 64 * 4);
  float* t64c  = (float*)alloc((size_t)BN_ * 64 * 4);
  float* g0    = (float*)alloc((size_t)BN_ * 256 * 4);   // gi (192 / 243)
  float* g1    = (float*)alloc((size_t)BN_ * 256 * 4);   // gh
  float* xw    = (float*)alloc((size_t)BN_ * 64 * 4);
  float* nk    = (float*)alloc((size_t)BN_ * 160 * 4);   // [gcn_out | pred_sigma | 0]
  float* nodeK = (float*)alloc((size_t)BN_ * 96 * 4);    // 81 valid + zero pad
  float* xrb   = (float*)alloc((size_t)BN_ * 16 * 4);
  float* crossb= (float*)alloc((size_t)BN_ * 9 * 4);
  float* phib  = (float*)alloc((size_t)BN_ * 9 * 4);

  auto gemm = [&](const float* A, int lda, const __bf16* PW, const float* bias,
                  float* C, int ldc, int K, int N, int act) {
    int Kt = (K + 31) / 32, Nt = (N + 15) / 16;
    long waves = (long)(BN_ / 16) * Nt;
    int blocks = (int)((waves * 32 + 255) / 256);
    gemm_wmma<<<blocks, 256, 0, stream>>>(A, lda, PW, bias, C, ldc, N, Kt, Nt, act);
  };

  // 1. node preprocessing (also writes x_tt1 output)
  node_prep<<<BN_ / 256, 256, 0, stream>>>(x11, x12, x22, Fm, Hm, meas, dyin,
                                           o_xtt1, rin, yfeat, xfeat);

  // 2. fc_dyi -> gnn[:, :64]
  gemm(yfeat, 32, p_dyi1, dyiB1, t64b, 64, 10, 64, 1);
  gemm(t64b, 64, p_dyi2, dyiB2, gnn, 128, 64, 64, 1);

  // 3. fc_r -> r_in
  gemm(rin, 32, p_r1, rB1, t64b, 64, 1, 64, 1);
  gemm(t64b, 64, p_r2, rB2, t64c, 64, 64, 64, 1);

  // 4. r_gru: gates then combine -> hidden_r_new (output) + gnn[:, 64:]
  gemm(t64c, 64, p_gih, gBih, g0, 256, 64, 192, 0);
  gemm(hr,   64, p_ghh, gBhh, g1, 256, 64, 192, 0);
  gru_combine<<<(BN_ * 64 + 255) / 256, 256, 0, stream>>>(g0, g1, hr, 64, o_hr, gnn,
                                                          64, 256, 128, 64, BN_ * 64);

  // 5. GCN: xw = gnn @ Wg ; mean-aggregate (uniform degree) + bias -> nk[:, :64]
  gemm(gnn, 128, p_gcn, nullptr, xw, 64, 128, 64, 0);
  gcn_agg<<<(BN_ * 64 + 255) / 256, 256, 0, stream>>>(xw, src, gcnB, nk);
  copy_ps<<<(BN_ * 96 + 255) / 256, 256, 0, stream>>>(psig, nk);

  // 6. fc_node -> node_kalman (K matrices), zero-padded to 96-wide rows
  gemm(nk, 160, p_n1, nB1, gnn, 128, 145, 128, 1);
  gemm(gnn, 128, p_n2, nB2, t64b, 64, 128, 64, 1);
  gemm(t64b, 64, p_n3, nB3, nodeK, 96, 64, 81, 0);

  // 7. fc_sig -> dxf; sigma_gru(dxf, node_kalman) -> pred_sigma_new (output)
  gemm(xfeat, 32, p_sig1, sigB1, t64c, 64, 18, 64, 1);
  gemm(t64c, 64, p_sig2, sigB2, t64b, 64, 64, 64, 1);
  gemm(t64b, 64, p_sgih, sgBih, g0, 256, 64, 243, 0);
  gemm(nodeK, 96, p_sghh, sgBhh, g1, 256, 81, 243, 0);
  gru_combine<<<(BN_ * O_ + 255) / 256, 256, 0, stream>>>(g0, g1, nodeK, 96, o_ps, nullptr,
                                                          81, 256, 0, 0, BN_ * O_);

  // 8. cross gain: xr = r_out @ Wr + br ; fused 5-layer edge MLP -> cross
  gemm(o_hr, 64, p_xr, crB, xrb, 16, 64, 9, 0);
  edge_mlp<<<BN_ / 8, 256, 0, stream>>>(xrb, Hm, meas, o_xtt1, src,
                                        p_c1, cB1, p_c2, cB2, p_c3, cB3,
                                        p_c4, cB4, p_c5, cB5, crossb);

  // 9. phi = x_tt1 + K @ cross ; SimpleConv mean -> x_tt (output)
  phi_kernel<<<(BN_ + 255) / 256, 256, 0, stream>>>(o_xtt1, nodeK, crossb, phib);
  xtt_kernel<<<(BN_ * S_ + 255) / 256, 256, 0, stream>>>(phib, src, o_xtt);

  // 10. pass-through edge_index output (raw int bits)
  hipMemcpyAsync(o_eidx, eidx, (size_t)2 * E_ * sizeof(int),
                 hipMemcpyDeviceToDevice, stream);
}
